// HarmonicRNN_24842090840504
// MI455X (gfx1250) — matrix operand order repaired and checked by reading, not compiled, not run on hardware
//
#include <hip/hip_runtime.h>
#include <hip/hip_bf16.h>
#include <math.h>

// ---------------------------------------------------------------------------
// HarmonicRNN on gfx1250 (wave32, WMMA 16x16x32 bf16, TDM + async-to-LDS)
//   K1: encoder input GEMMs — weights via TENSOR_LOAD_TO_LDS (TDM),
//       activations via GLOBAL_LOAD_ASYNC_TO_LDS_B128, bf16 WMMA compute
//   K2: 8-step GRU recurrence, one persistent WGP per direction
//   K3: hidden-state flatten quirk + decoder state init
//   K4: 128-step autoregressive decoder, 24 persistent blocks + grid barrier
// ---------------------------------------------------------------------------

typedef __attribute__((ext_vector_type(16))) __bf16 v16bf;
typedef __attribute__((ext_vector_type(8)))  float  v8f;
typedef __attribute__((ext_vector_type(4)))  unsigned int v4u;
typedef __attribute__((ext_vector_type(8)))  int    v8i;
typedef __attribute__((ext_vector_type(4)))  int    v4i;

#define SR   16000
#define TT   8
#define BB   16
#define H2   512
#define HID  1024
#define G3E  1536   // 3*H2
#define G3D  3072   // 3*HID
#define DEC_BLOCKS 24

__device__ __forceinline__ float sigm(float x) { return 1.0f / (1.0f + __expf(-x)); }

// Convert two groups of 8 contiguous fp32 into one bf16 WMMA fragment register set.
__device__ __forceinline__ v16bf cvt16(const float* a, const float* b) {
  v16bf r;
#pragma unroll
  for (int i = 0; i < 8; ++i) { r[i] = (__bf16)a[i]; r[i + 8] = (__bf16)b[i]; }
  return r;
}
// A fragment (16x32 bf16): lane holds row m=lane%16, half=lane/16.
// Element e maps to k = e + half*16  ->  two contiguous 8-float runs.
__device__ __forceinline__ v16bf fragA(const float* row32, int half) {
  return cvt16(row32 + half * 16, row32 + half * 16 + 8);
}
// B fragment (32x16 bf16): lane holds col n=lane%16, khalf=lane/16; k = e + khalf*16.
__device__ __forceinline__ v16bf fragB(const float* p16) {
  return cvt16(p16, p16 + 8);
}
__device__ __forceinline__ v8f wmma_bf16(v16bf a, v16bf b, v8f c) {
  // (neg_a, A, neg_b, B, c_mod, C, reuse_a, reuse_b)
  return __builtin_amdgcn_wmma_f32_16x16x32_bf16(false, a, false, b, (short)0, c,
                                                 false, false);
}

__device__ __forceinline__ unsigned lds_addr_of(const void* p) {
  return (unsigned)(uintptr_t)p;   // low 32 bits of flat shared address = LDS offset
}

// TDM: DMA a 2D tile (tile_d1 rows x tile_d0 fp32 elems, row stride = stride
// elems) from global memory into contiguous LDS at lds_byte. D# per ISA §8.3/8.4:
// group0 = {count=1 | lds_addr | global_addr | type=2}, group1 packs data_size=4B,
// tensor dims, tile dims, dim0 stride. 2D tensor -> groups 2/3 zero (disabled).
__device__ __forceinline__ void tdm_load_tile(const float* gsrc, unsigned lds_byte,
                                              int tensor_d0, int tensor_d1,
                                              int tile_d0, int tile_d1,
                                              int stride_elems) {
  unsigned long long ga = (unsigned long long)(uintptr_t)gsrc;
  v4u g0;
  g0[0] = 1u;                                               // count=1, user mode
  g0[1] = lds_byte;                                         // lds_addr
  g0[2] = (unsigned)(ga & 0xFFFFFFFFu);                     // global_addr[31:0]
  g0[3] = (unsigned)((ga >> 32) & 0x1FFFFFFu) | (2u << 30); // addr[56:32] | type=2
  v8i g1;
  g1[0] = (int)(2u << 16);                                  // data_size=2 (4B)
  g1[1] = (int)(((unsigned)tensor_d0 & 0xFFFFu) << 16);     // tensor_dim0[15:0]
  g1[2] = (int)((((unsigned)tensor_d0 >> 16) & 0xFFFFu) |
                (((unsigned)tensor_d1 & 0xFFFFu) << 16));   // dim0 hi | dim1 lo
  g1[3] = (int)((((unsigned)tensor_d1 >> 16) & 0xFFFFu) |
                ((unsigned)tile_d0 << 16));                 // dim1 hi | tile_dim0
  g1[4] = (int)((unsigned)tile_d1 & 0xFFFFu);               // tile_dim1, tile_dim2=0
  g1[5] = stride_elems;                                     // tensor_dim0_stride lo
  g1[6] = 0;                                                // stride hi | dim1_stride
  g1[7] = 0;
  v4i z4 = {};
#if defined(__clang_major__) && (__clang_major__ >= 23)
  v8i z8 = {};
  __builtin_amdgcn_tensor_load_to_lds(g0, g1, z4, z4, z8, 0);
#else
  __builtin_amdgcn_tensor_load_to_lds(g0, g1, z4, z4, 0);
#endif
}

// ---------------------------------------------------------------------------
// K1: gi[dir][r=t*16+b][g] = sum_s x[b, t(or 7-t), s] * wi[g,s] + bi[g]
// Grid (48 col-blocks of 32 g, 2 dirs) x 256 threads (8 waves).
// Wave w: N-tile nt=w&1, M-tiles {(w>>1)*2, +1}; each weight loaded once overall.
// W tile arrives via TDM (TENSORcnt), A tile via async-to-LDS (ASYNCcnt).
// ---------------------------------------------------------------------------
__global__ __launch_bounds__(256) void k_enc_in(
    const float* __restrict__ x,
    const float* __restrict__ wi_f, const float* __restrict__ bi_f,
    const float* __restrict__ wi_b, const float* __restrict__ bi_b,
    float* __restrict__ gi_f, float* __restrict__ gi_b)
{
  const int dir = blockIdx.y;
  const int cb  = blockIdx.x;
  const float* wi = dir ? wi_b : wi_f;
  const float* bi = dir ? bi_b : bi_f;
  float*       gi = dir ? gi_b : gi_f;

  __shared__ __align__(16) float As[128 * 32];
  __shared__ __align__(16) float Ws[32 * 32];

  const int tid  = threadIdx.x;
  const int w    = tid >> 5, lane = tid & 31;
  const int half = lane >> 4, n16 = lane & 15;
  const int nt   = w & 1;
  const int mt0  = (w >> 1) * 2;

  // A staging: thread owns half a row (16 floats = 4x b128) of the 128x32 chunk
  const int lr   = tid >> 1;
  const int soff = (tid & 1) * 16;
  const int ta = lr >> 4, ba = lr & 15;
  const int tmem = dir ? (7 - ta) : ta;
  const unsigned aRowByte = (unsigned)((size_t)(ba * TT + tmem) * SR * 4u);
  const float* aRow = x + (size_t)(ba * TT + tmem) * SR;
  const float* wBase = wi + (size_t)(cb * 32) * SR;
  const unsigned asLds = lds_addr_of(As) + (unsigned)(lr * 32 + soff) * 4u;
  const unsigned wsLds = lds_addr_of(Ws);

  v8f acc0 = {}, acc1 = {};

  for (int kc = 0; kc < SR / 32; ++kc) {
    const int k0 = kc * 32;
    // ---- A tile: GLOBAL_LOAD_ASYNC_TO_LDS_B128 (GVS: saddr=x, vaddr=byte off)
    {
      unsigned voff = aRowByte + (unsigned)(k0 + soff) * 4u;
#pragma unroll
      for (int i = 0; i < 4; ++i) {
        asm volatile("global_load_async_to_lds_b128 %0, %1, %2"
                     :
                     : "v"(asLds + 16u * i), "v"(voff + 16u * i), "s"(x)
                     : "memory");
      }
    }
    // ---- W tile: Tensor Data Mover, 32x32 fp32 tile, row stride 16000
    if (w == 0) {
      tdm_load_tile(wBase + k0, wsLds,
                    /*tensor_d0=*/SR - k0, /*tensor_d1=*/G3E - cb * 32,
                    /*tile_d0=*/32, /*tile_d1=*/32, /*stride=*/SR);
    }
    if (kc + 1 < SR / 32)                     // global_prefetch_b8 next A chunk
      __builtin_prefetch(aRow + k0 + 32, 0, 0);

    asm volatile("s_wait_asynccnt 0x0" ::: "memory");
    __builtin_amdgcn_s_wait_tensorcnt(0);
    __syncthreads();

    v16bf bfr = fragB(Ws + (nt * 16 + n16) * 32 + half * 16);
    v16bf a0  = fragA(As + (mt0 * 16 + n16) * 32, half);
    v16bf a1  = fragA(As + ((mt0 + 1) * 16 + n16) * 32, half);
    acc0 = wmma_bf16(a0, bfr, acc0);
    acc1 = wmma_bf16(a1, bfr, acc1);
    __syncthreads();
  }

  const int g = cb * 32 + nt * 16 + n16;
  const float bv = bi[g];
#pragma unroll
  for (int d = 0; d < 8; ++d) {
    int r0 = mt0 * 16 + d + half * 8;       // C/D layout: m = d + (lane>>4)*8
    gi[(size_t)r0 * G3E + g]        = acc0[d] + bv;
    gi[(size_t)(r0 + 16) * G3E + g] = acc1[d] + bv;
  }
}

// ---------------------------------------------------------------------------
// K2: 8-step GRU scan per direction; one block (512 thr = 16 waves) per dir.
// h lives in LDS; gh round-trips through a per-dir global scratch (192 KB).
// ---------------------------------------------------------------------------
__global__ __launch_bounds__(512) void k_enc_rnn(
    const float* __restrict__ gi_f, const float* __restrict__ gi_b,
    const float* __restrict__ wh_f, const float* __restrict__ bh_f,
    const float* __restrict__ wh_b, const float* __restrict__ bh_b,
    float* __restrict__ hf, float* __restrict__ hb, float* __restrict__ gh_ws)
{
  const int dir = blockIdx.x;
  const float* gi = dir ? gi_b : gi_f;
  const float* Wh = dir ? wh_b : wh_f;
  const float* bh = dir ? bh_b : bh_f;
  float* hout = dir ? hb : hf;
  float* gh = gh_ws + (size_t)dir * (BB * G3E);

  __shared__ __align__(16) float hl[BB * H2];
  const int tid = threadIdx.x;
  for (int i = tid; i < BB * H2; i += 512) hl[i] = 0.0f;
  __syncthreads();

  const int w = tid >> 5, lane = tid & 31;
  const int half = lane >> 4, n16 = lane & 15;

  for (int t = 0; t < TT; ++t) {
    v8f acc[6] = {};
    for (int kc = 0; kc < H2 / 32; ++kc) {
      v16bf a = fragA(hl + n16 * H2 + kc * 32, half);
#pragma unroll
      for (int j = 0; j < 6; ++j) {
        int g = (w * 6 + j) * 16 + n16;
        v16bf bf = fragB(Wh + (size_t)g * H2 + kc * 32 + half * 16);
        acc[j] = wmma_bf16(a, bf, acc[j]);
      }
    }
#pragma unroll
    for (int j = 0; j < 6; ++j) {
      int g = (w * 6 + j) * 16 + n16;
      float bv = bh[g];
#pragma unroll
      for (int d = 0; d < 8; ++d)
        gh[(size_t)(d + half * 8) * G3E + g] = acc[j][d] + bv;
    }
    __syncthreads();

    const float* girow = gi + (size_t)(t * BB) * G3E;
    for (int i = tid; i < BB * H2; i += 512) {
      int b = i >> 9, j = i & (H2 - 1);
      float ir = girow[(size_t)b * G3E + j];
      float iz = girow[(size_t)b * G3E + j + H2];
      float in = girow[(size_t)b * G3E + j + 2 * H2];
      float hr = gh[(size_t)b * G3E + j];
      float hz = gh[(size_t)b * G3E + j + H2];
      float hn = gh[(size_t)b * G3E + j + 2 * H2];
      float r = sigm(ir + hr), z = sigm(iz + hz);
      float n = tanhf(in + r * hn);
      hl[i] = (1.0f - z) * n + z * hl[i];
    }
    __syncthreads();
  }
  for (int i = tid; i < BB * H2; i += 512) hout[i] = hl[i];
}

// ---------------------------------------------------------------------------
// K3: hidden0 = stack([hf,hb]).reshape(B,1024) flatten quirk + state init
// ---------------------------------------------------------------------------
__global__ __launch_bounds__(256) void k_init_dec(
    const float* __restrict__ hf, const float* __restrict__ hb,
    float* __restrict__ h, float* __restrict__ last, int* __restrict__ st)
{
  const int tid = threadIdx.x;
  for (int i = tid; i < BB * HID; i += 256) {
    int d = i >> 13, bb2 = (i >> 9) & 15, k = i & 511;
    h[i] = d ? hb[bb2 * H2 + k] : hf[bb2 * H2 + k];
  }
  if (tid < 96) last[tid] = 0.0f;
  if (tid < 16) { st[tid] = 0; st[16 + tid] = 1; st[32 + tid] = 0; }  // msgs, active, cond
  if (tid == 0) { st[48] = 0; st[49] = 0; }                           // barrier cnt, gen
}

// ---------------------------------------------------------------------------
// K4: persistent decoder. 24 blocks x 256 thr = 192 waves = 192 N-tiles of gh.
// ---------------------------------------------------------------------------
__device__ __forceinline__ void grid_barrier(int* cnt, int* gen, int nblk) {
  __syncthreads();
  if (threadIdx.x == 0) {
    __threadfence();
    int g = __hip_atomic_load(gen, __ATOMIC_RELAXED, __HIP_MEMORY_SCOPE_AGENT);
    int a = __hip_atomic_fetch_add(cnt, 1, __ATOMIC_ACQ_REL, __HIP_MEMORY_SCOPE_AGENT);
    if (a == nblk - 1) {
      __hip_atomic_store(cnt, 0, __ATOMIC_RELAXED, __HIP_MEMORY_SCOPE_AGENT);
      __hip_atomic_fetch_add(gen, 1, __ATOMIC_RELEASE, __HIP_MEMORY_SCOPE_AGENT);
    } else {
      while (__hip_atomic_load(gen, __ATOMIC_ACQUIRE, __HIP_MEMORY_SCOPE_AGENT) == g)
        __builtin_amdgcn_s_sleep(2);
    }
  }
  __syncthreads();
}

__global__ __launch_bounds__(256) void k_decode(
    float* __restrict__ h, float* __restrict__ gh,
    float* __restrict__ last, int* __restrict__ st,
    const float* __restrict__ dwi, const float* __restrict__ dwh,
    const float* __restrict__ dbi, const float* __restrict__ dbh,
    const float* __restrict__ lw, const float* __restrict__ lb,
    float* __restrict__ out)
{
  int* msgs = st; int* active = st + 16; int* cond = st + 32;
  int* cnt = st + 48; int* gen = st + 49;
  const int tid = threadIdx.x;
  const int w = tid >> 5, lane = tid & 31;
  const int half = lane >> 4, n16 = lane & 15;
  const int wg = blockIdx.x * 8 + w;     // 0..191 == N-tile id

  for (int step = 0; step < 128; ++step) {
    // stage 1: gh = h @ dwh^T + dbh   (M=16 batch, N=3072, K=1024, bf16 WMMA)
    {
      v8f acc = {};
      const int g = wg * 16 + n16;
      for (int kc = 0; kc < HID / 32; ++kc) {
        v16bf a  = fragA(h + (size_t)n16 * HID + kc * 32, half);
        v16bf bf = fragB(dwh + (size_t)g * HID + kc * 32 + half * 16);
        acc = wmma_bf16(a, bf, acc);
      }
      float bv = dbh[g];
#pragma unroll
      for (int d = 0; d < 8; ++d)
        gh[(size_t)(d + half * 8) * G3D + g] = acc[d] + bv;
    }
    grid_barrier(cnt, gen, DEC_BLOCKS);

    // stage 3: GRU elementwise; gi from tiny (3072x6) dwi; commit h if cond
    for (int i = blockIdx.x * 256 + tid; i < BB * HID; i += DEC_BLOCKS * 256) {
      int b = i >> 10, j = i & (HID - 1);
      bool cnd = (active[b] != 0) && (rintf(last[b * 6]) != 8.0f) && (msgs[b] < 127);
      float l[6];
#pragma unroll
      for (int c = 0; c < 6; ++c) l[c] = last[b * 6 + c];
      float giv[3];
#pragma unroll
      for (int s = 0; s < 3; ++s) {
        int g = j + s * HID;
        float v = dbi[g];
#pragma unroll
        for (int c = 0; c < 6; ++c) v += l[c] * dwi[g * 6 + c];
        giv[s] = v;
      }
      float hr = gh[(size_t)b * G3D + j];
      float hz = gh[(size_t)b * G3D + j + HID];
      float hn = gh[(size_t)b * G3D + j + 2 * HID];
      float r = sigm(giv[0] + hr), z = sigm(giv[1] + hz);
      float n = tanhf(giv[2] + r * hn);
      float hnew = (1.0f - z) * n + z * h[(size_t)b * HID + j];
      if (cnd) h[(size_t)b * HID + j] = hnew;
      if (j == 0) cond[b] = cnd ? 1 : 0;
    }
    grid_barrier(cnt, gen, DEC_BLOCKS);

    // stage 5: out = h @ lw^T + lb; midi, last, msgs, active  (96 (b,c) pairs)
    if (wg < 96) {
      int b = wg / 6, c = wg % 6;
      float s = 0.0f;
      for (int j = lane; j < HID; j += 32)
        s += h[(size_t)b * HID + j] * lw[c * HID + j];
#pragma unroll
      for (int m = 16; m >= 1; m >>= 1) s += __shfl_xor(s, m, 32);  // wave32
      if (lane == 0) {
        int cnd = cond[b];
        float o = s + lb[c];
        out[((size_t)b * 128 + step) * 6 + c] = cnd ? o : 0.0f;
        if (cnd) last[b * 6 + c] = o;
      }
    }
    if (blockIdx.x == 12 && tid < 16) {
      int cnd = cond[tid];
      if (cnd) msgs[tid] += 1;
      active[tid] = cnd;
    }
    grid_barrier(cnt, gen, DEC_BLOCKS);
  }

  if (blockIdx.x == 12 && tid < 16)
    out[12288 + tid] = (float)(msgs[tid] + 1);   // num_messages
}

// ---------------------------------------------------------------------------
extern "C" void kernel_launch(void* const* d_in, const int* in_sizes, int n_in,
                              void* d_out, int out_size, void* d_ws, size_t ws_size,
                              hipStream_t stream) {
  (void)in_sizes; (void)n_in; (void)out_size; (void)ws_size;
  const float* x    = (const float*)d_in[0];
  const float* wi_f = (const float*)d_in[1];
  const float* wh_f = (const float*)d_in[2];
  const float* bi_f = (const float*)d_in[3];
  const float* bh_f = (const float*)d_in[4];
  const float* wi_b = (const float*)d_in[5];
  const float* wh_b = (const float*)d_in[6];
  const float* bi_b = (const float*)d_in[7];
  const float* bh_b = (const float*)d_in[8];
  const float* dwi  = (const float*)d_in[9];
  const float* dwh  = (const float*)d_in[10];
  const float* dbi  = (const float*)d_in[11];
  const float* dbh  = (const float*)d_in[12];
  const float* lw   = (const float*)d_in[13];
  const float* lb   = (const float*)d_in[14];

  float* ws   = (float*)d_ws;                 // ~2.1 MB of scratch
  float* gi_f = ws;
  float* gi_b = gi_f + 128 * G3E;
  float* hf   = gi_b + 128 * G3E;
  float* hb   = hf + BB * H2;
  float* hdec = hb + BB * H2;
  float* ghe  = hdec + BB * HID;
  float* ghd  = ghe + 2 * BB * G3E;
  float* last = ghd + BB * G3D;
  int*   st   = (int*)(last + 96);

  k_enc_in <<<dim3(48, 2), 256, 0, stream>>>(x, wi_f, bi_f, wi_b, bi_b, gi_f, gi_b);
  k_enc_rnn<<<2, 512, 0, stream>>>(gi_f, gi_b, wh_f, bh_f, wh_b, bh_b, hf, hb, ghe);
  k_init_dec<<<1, 256, 0, stream>>>(hf, hb, hdec, last, st);
  k_decode <<<DEC_BLOCKS, 256, 0, stream>>>(hdec, ghd, last, st,
                                            dwi, dwh, dbi, dbh, lw, lb,
                                            (float*)d_out);
}
